// LorentzMultiHeadedAttention_4097398801027
// MI455X (gfx1250) — matrix operand order repaired and checked by reading, not compiled, not run on hardware
//
#include <hip/hip_runtime.h>
#include <cmath>

// ---------------------------------------------------------------------------
// Lorentz multi-head attention for MI455X (gfx1250, wave32, WMMA).
// All matmuls use v_wmma_f32_16x16x32_bf16 with a bf16x3 split
// (hi*hi + hi*lo + lo*hi, f32 accumulate) for ~f32 accuracy at bf16 speed.
// Operands that are reused (W, K, V, Q) are pre-split once into bf16 hi/lo
// arrays so the hot WMMA loops issue pure vector loads, no conversion VALU.
// ---------------------------------------------------------------------------

#define BATCH 4
#define SEQ   1024
#define DIM   1024
#define NHEAD 16
#define DHEAD 64

typedef __attribute__((ext_vector_type(16))) __bf16 v16bf;
typedef __attribute__((ext_vector_type(8)))  float  v8f;

__device__ __forceinline__ v8f wmma_bf16x3(const v16bf ah, const v16bf al,
                                           const v16bf bh, const v16bf bl, v8f c) {
  c = __builtin_amdgcn_wmma_f32_16x16x32_bf16(false, ah, false, bh, (short)0, c, false, false);
  c = __builtin_amdgcn_wmma_f32_16x16x32_bf16(false, ah, false, bl, (short)0, c, false, false);
  c = __builtin_amdgcn_wmma_f32_16x16x32_bf16(false, al, false, bh, (short)0, c, false, false);
  return c;
}

__device__ __forceinline__ void split16(const float* f, v16bf& h, v16bf& l) {
#pragma unroll
  for (int i = 0; i < 16; ++i) {
    float x = f[i];
    __bf16 hi = (__bf16)x;
    h[i] = hi;
    l[i] = (__bf16)(x - (float)hi);
  }
}

// A-fragment (16x32 bf16, MxK). Lane L holds row M = L&15.
// Lanes 0-15: K = 0..15 ; lanes 16-31: K = 16..31.
__device__ __forceinline__ void load_afrag_f32(const float* rowp, int kb, v16bf& h, v16bf& l) {
  float buf[16];
#pragma unroll
  for (int i = 0; i < 16; ++i) buf[i] = rowp[kb + i];
  split16(buf, h, l);
}

// A-fragment direct from a pre-split bf16 row.
__device__ __forceinline__ v16bf load_afrag_bf(const __bf16* rowp, int kb) {
  v16bf v;
#pragma unroll
  for (int i = 0; i < 16; ++i) v[i] = rowp[kb + i];
  return v;
}

// B-fragment (32x16 bf16, KxN). Lane L holds column N = L&15; 16 contiguous K
// per lane half (kb2 = 0/16). ptr must be 32B aligned (it is, by layout).
__device__ __forceinline__ v16bf load_bfrag_bf(const __bf16* p) {
  return *reinterpret_cast<const v16bf*>(p);
}

// ---------------------------------------------------------------------------
// Kernel 0: pre-split Wk/Wv/Wq (f32 -> bf16 hi/lo), concatenated per z.
// ---------------------------------------------------------------------------
__global__ __launch_bounds__(256) void wsplit_kernel(
    const float* __restrict__ Wk, const float* __restrict__ Wv, const float* __restrict__ Wq,
    __bf16* __restrict__ Wh, __bf16* __restrict__ Wl)
{
  const int z = blockIdx.y;
  const float* W = (z == 0) ? Wk : (z == 1 ? Wv : Wq);
  const size_t i = ((size_t)blockIdx.x * 256 + threadIdx.x) * 4;
  const float4 x = *reinterpret_cast<const float4*>(W + i);
  const size_t o = (size_t)z * DIM * DIM + i;
  const float xs[4] = {x.x, x.y, x.z, x.w};
#pragma unroll
  for (int j = 0; j < 4; ++j) {
    __bf16 hi = (__bf16)xs[j];
    Wh[o + j] = hi;
    Wl[o + j] = (__bf16)(xs[j] - (float)hi);
  }
}

// ---------------------------------------------------------------------------
// Kernel 1: projections  Y = X @ W^T  (+ Lorentz nonlinearity), z selects K/V/Q.
// Block = 128 threads (4 waves); each wave owns a 16x64 output tile = 1 head.
// Outputs stored as bf16 hi/lo pairs: K,Q as (b,h,s,d) (Q time pre-negated),
// V transposed as (b,h,d,s) so attention AV B-fragments are contiguous.
// ---------------------------------------------------------------------------
__global__ __launch_bounds__(128) void lorentz_proj_kernel(
    const float* __restrict__ key, const float* __restrict__ value, const float* __restrict__ query,
    const __bf16* __restrict__ Wh, const __bf16* __restrict__ Wl,
    const float* __restrict__ sk,  const float* __restrict__ sv,  const float* __restrict__ sq,
    __bf16* __restrict__ Kh, __bf16* __restrict__ Kl,
    __bf16* __restrict__ Vh, __bf16* __restrict__ Vl,
    __bf16* __restrict__ Qh, __bf16* __restrict__ Ql)
{
  __shared__ float lds[64 * 64];
  const int z = blockIdx.z;
  const float* X  = (z == 0) ? key : (z == 1 ? value : query);
  const float  ls = (z == 0) ? sk[0] : (z == 1 ? sv[0] : sq[0]);
  const __bf16* Whz = Wh + (size_t)z * DIM * DIM;
  const __bf16* Wlz = Wl + (size_t)z * DIM * DIM;

  const int wave = threadIdx.x >> 5;
  const int lane = threadIdx.x & 31;
  const int M0   = blockIdx.x * 64 + wave * 16;   // global row (b*SEQ + s)
  const int h    = blockIdx.y;                    // head = 64-col tile
  const int N0   = h * DHEAD;
  const int m    = lane & 15;
  const int kb   = (lane < 16) ? 0 : 8;
  const int kb2  = (lane < 16) ? 0 : 16;
  const int rowoff = (lane >> 4) * 8;

  v8f acc[4] = {};
  const float* xrow = X + (size_t)(M0 + m) * DIM;
  for (int k0 = 0; k0 < DIM; k0 += 32) {
    v16bf ah, al;
    load_afrag_f32(xrow + k0, kb2, ah, al);
#pragma unroll
    for (int nt = 0; nt < 4; ++nt) {
      const size_t woff = (size_t)(N0 + nt * 16 + m) * DIM + k0 + kb2;
      acc[nt] = wmma_bf16x3(ah, al, load_bfrag_bf(Whz + woff), load_bfrag_bf(Wlz + woff), acc[nt]);
    }
  }

  // stage C fragments (row r+rowoff, col nt*16+m) into this wave's LDS slab
  float* myl = lds + wave * 16 * 64;
#pragma unroll
  for (int nt = 0; nt < 4; ++nt)
#pragma unroll
    for (int r = 0; r < 8; ++r)
      myl[(rowoff + r) * 64 + nt * 16 + m] = acc[nt][r];
  __syncthreads();

  if (lane < 16) {
    const float* y = myl + lane * 64;
    const float t  = (1.0f / (1.0f + expf(-y[0]))) * expf(ls) + 1.1f;  // sigmoid*e^ls+1.1
    float ssq = 0.0f;
#pragma unroll
    for (int d = 1; d < 64; ++d) ssq += y[d] * y[d];
    const float sc = sqrtf((t * t - 1.0f) / ssq);   // K_CURV = 1
    const int gm = M0 + lane;
    const int bb = gm >> 10;          // / SEQ
    const int sr = gm & (SEQ - 1);    // % SEQ
    float v[64];
    v[0] = (z == 2) ? -t : t;         // Q: minkowski-flip the time component now
#pragma unroll
    for (int d = 1; d < 64; ++d) v[d] = y[d] * sc;
    if (z == 1) {                     // V transposed: (b,h,d,s)
      const size_t base = ((size_t)(bb * NHEAD + h) * DHEAD) * SEQ + sr;
#pragma unroll
      for (int d = 0; d < 64; ++d) {
        __bf16 hi = (__bf16)v[d];
        Vh[base + (size_t)d * SEQ] = hi;
        Vl[base + (size_t)d * SEQ] = (__bf16)(v[d] - (float)hi);
      }
    } else {                          // K / Q: (b,h,s,d)
      __bf16* dh = ((z == 0) ? Kh : Qh) + ((size_t)(bb * NHEAD + h) * SEQ + sr) * DHEAD;
      __bf16* dl = ((z == 0) ? Kl : Ql) + ((size_t)(bb * NHEAD + h) * SEQ + sr) * DHEAD;
#pragma unroll
      for (int d = 0; d < 64; ++d) {
        __bf16 hi = (__bf16)v[d];
        dh[d] = hi;
        dl[d] = (__bf16)(v[d] - (float)hi);
      }
    }
  }
}

// ---------------------------------------------------------------------------
// Kernel 2: flash-style attention, one wave per (b, h, 16-query tile).
// ---------------------------------------------------------------------------
__global__ __launch_bounds__(32) void lorentz_attn_kernel(
    const __bf16* __restrict__ Qh, const __bf16* __restrict__ Ql,
    const __bf16* __restrict__ Kh, const __bf16* __restrict__ Kl,
    const __bf16* __restrict__ Vh, const __bf16* __restrict__ Vl,
    const int* __restrict__ mask, const float* __restrict__ ascale_p,
    const float* __restrict__ abias_p, float* __restrict__ out)
{
  __shared__ float lds[16 * 64];      // S/P tile (16x32) in loop; O staging (16x64) at end
  __shared__ float rowscale[16];
  const int lane = threadIdx.x;
  const int qt = blockIdx.x, h = blockIdx.y, b = blockIdx.z;
  const int m     = lane & 15;
  const int kb    = (lane < 16) ? 0 : 8;
  const int kb2   = (lane < 16) ? 0 : 16;
  const int rowoff = (lane >> 4) * 8;
  const int jbase  = (lane >> 4) * 16;     // this lane's 16-column half of its row

  const float ascale = ascale_p[0];
  const float abias  = abias_p[0];
  const float c2     = 2.0f / ascale;

  const size_t qoff  = ((size_t)((b * NHEAD + h) * SEQ) + qt * 16 + m) * DHEAD;
  const __bf16* Kbh  = Kh + (size_t)((b * NHEAD + h) * SEQ) * DHEAD;
  const __bf16* Kbl  = Kl + (size_t)((b * NHEAD + h) * SEQ) * DHEAD;
  const __bf16* Vbh  = Vh + (size_t)((b * NHEAD + h) * DHEAD) * SEQ;
  const __bf16* Vbl  = Vl + (size_t)((b * NHEAD + h) * DHEAD) * SEQ;

  // Q fragments (time already negated): 16x64 = two 16x32 A-frags, hi & lo
  v16bf qfh[2], qfl[2];
#pragma unroll
  for (int ks = 0; ks < 2; ++ks) {
    qfh[ks] = load_afrag_bf(Qh + qoff + ks * 32, kb2);
    qfl[ks] = load_afrag_bf(Ql + qoff + ks * 32, kb2);
  }

  v8f o[4] = {};
  float rmax = -__builtin_inff(), lrow = 0.0f;

  // Gauss-weight row constants (width = 0.5/65 -> 1/(2w^2) = 8450 exactly;
  // the INV_SQRT_2PI/width prefactor cancels under row-max normalization).
  const int   qg     = qt * 16 + m;
  const float center = (float)qg * (1.0f / 1024.0f);
  const float kstar  = rintf((float)qg * (1023.0f / 1024.0f));
  const float dmin   = kstar * (1.0f / 1023.0f) - center;
  const float gb     = dmin * dmin;
  const float inv2w2 = 8450.0f;

  for (int kt = 0; kt < SEQ / 32; ++kt) {
    const int n0 = kt * 32;

    // ---- S = Q' K^T  (two 16x16 subtiles, K=64 => 2 k-steps, bf16x3 each) ----
    v8f s_c[2] = {};
#pragma unroll
    for (int st = 0; st < 2; ++st) {
      const size_t kcol = (size_t)(n0 + st * 16 + m) * DHEAD + kb2;
#pragma unroll
      for (int ks = 0; ks < 2; ++ks)
        s_c[st] = wmma_bf16x3(qfh[ks], qfl[ks],
                              load_bfrag_bf(Kbh + kcol + ks * 32),
                              load_bfrag_bf(Kbl + kcol + ks * 32), s_c[st]);
    }
#pragma unroll
    for (int st = 0; st < 2; ++st)
#pragma unroll
      for (int r = 0; r < 8; ++r)
        lds[(rowoff + r) * 32 + st * 16 + m] = s_c[st][r];
    __syncthreads();

    // ---- per-row: affine, gauss weight, mask, online softmax ----
    // Row m is handled by lanes m and m+16 (16 columns each), combined by shuffle.
    {
      float sbuf[16];
      float bmax = -1e30f;
      const int* mp = mask + (size_t)(b * SEQ + qg) * SEQ + n0 + jbase;
#pragma unroll
      for (int j = 0; j < 16; ++j) {
        float s = lds[m * 32 + jbase + j];
        s = c2 * s + (c2 + abias);                        // (2 + 2<q',k>)/scale + bias
        const float dd = (float)(n0 + jbase + j) * (1.0f / 1023.0f) - center;
        s *= expf((gb - dd * dd) * inv2w2);               // normalized gauss weight
        if (mp[j] == 0) s = -1e18f;
        sbuf[j] = s;
        bmax = fmaxf(bmax, s);
      }
      bmax = fmaxf(bmax, __shfl_xor(bmax, 16, 32));
      const float mnew = fmaxf(rmax, bmax);
      const float rs   = expf(rmax - mnew);               // first iter: exp(-inf)=0
      float bsum = 0.0f;
#pragma unroll
      for (int j = 0; j < 16; ++j) {
        const float p = expf(sbuf[j] - mnew);
        lds[m * 32 + jbase + j] = p;
        bsum += p;
      }
      bsum += __shfl_xor(bsum, 16, 32);
      lrow = lrow * rs + bsum;
      rmax = mnew;
      if (lane < 16) rowscale[m] = rs;
    }
    __syncthreads();

    // ---- rescale O, then O += P @ V ----
    float rs8[8];
#pragma unroll
    for (int r = 0; r < 8; ++r) rs8[r] = rowscale[rowoff + r];
#pragma unroll
    for (int nt = 0; nt < 4; ++nt)
#pragma unroll
      for (int r = 0; r < 8; ++r) o[nt][r] *= rs8[r];

    float pbuf[16];
#pragma unroll
    for (int i = 0; i < 16; ++i) pbuf[i] = lds[m * 32 + kb2 + i];
    v16bf ph, pl;
    split16(pbuf, ph, pl);

#pragma unroll
    for (int nt = 0; nt < 4; ++nt) {
      const size_t vcol = (size_t)(nt * 16 + m) * SEQ + n0 + kb2;   // Vt: contiguous in key
      o[nt] = wmma_bf16x3(ph, pl, load_bfrag_bf(Vbh + vcol), load_bfrag_bf(Vbl + vcol), o[nt]);
    }
    __syncthreads();
  }

  // ---- finalize: mu = O/l, Minkowski normalize, write (b,s,h*64+d) ----
#pragma unroll
  for (int nt = 0; nt < 4; ++nt)
#pragma unroll
    for (int r = 0; r < 8; ++r)
      lds[(rowoff + r) * 64 + nt * 16 + m] = o[nt][r];
  __syncthreads();

  if (lane < 16) {
    const float linv = 1.0f / lrow;
    const float* orow = lds + lane * 64;
    float mu[64];
#pragma unroll
    for (int d = 0; d < 64; ++d) mu[d] = orow[d] * linv;
    float lnorm = -mu[0] * mu[0];
#pragma unroll
    for (int d = 1; d < 64; ++d) lnorm += mu[d] * mu[d];
    const float inv = 1.0f / sqrtf(fmaxf(fabsf(lnorm), 1e-8f));
    float* dst = out + (size_t)(b * SEQ + qg) * DIM + h * DHEAD;
#pragma unroll
    for (int d = 0; d < 64; ++d) dst[d] = mu[d] * inv;
  }
}

// ---------------------------------------------------------------------------
extern "C" void kernel_launch(void* const* d_in, const int* in_sizes, int n_in,
                              void* d_out, int out_size, void* d_ws, size_t ws_size,
                              hipStream_t stream) {
  const float* key    = (const float*)d_in[0];
  const float* value  = (const float*)d_in[1];
  const float* query  = (const float*)d_in[2];
  const int*   mask   = (const int*)  d_in[3];
  const float* Wk     = (const float*)d_in[4];
  const float* Wv     = (const float*)d_in[5];
  const float* Wq     = (const float*)d_in[6];
  const float* sk     = (const float*)d_in[7];
  const float* sv     = (const float*)d_in[8];
  const float* sq     = (const float*)d_in[9];
  const float* ascale = (const float*)d_in[10];
  const float* abias  = (const float*)d_in[11];
  float* out = (float*)d_out;

  // workspace (bf16): K,V,Q hi/lo pairs (6 x 8 MB) + split W (2 x 6 MB) = 60 MB
  const size_t pe = (size_t)BATCH * NHEAD * SEQ * DHEAD;   // 4 Mi elements
  __bf16* Kh = (__bf16*)d_ws;
  __bf16* Kl = Kh + pe;
  __bf16* Vh = Kl + pe;
  __bf16* Vl = Vh + pe;
  __bf16* Qh = Vl + pe;
  __bf16* Ql = Qh + pe;
  __bf16* Wh = Ql + pe;
  __bf16* Wl = Wh + (size_t)3 * DIM * DIM;

  dim3 gw(DIM * DIM / (256 * 4), 3), bw(256);
  wsplit_kernel<<<gw, bw, 0, stream>>>(Wk, Wv, Wq, Wh, Wl);

  dim3 gp(BATCH * SEQ / 64, NHEAD, 3), bp(128);
  lorentz_proj_kernel<<<gp, bp, 0, stream>>>(key, value, query, Wh, Wl,
                                             sk, sv, sq, Kh, Kl, Vh, Vl, Qh, Ql);

  dim3 ga(SEQ / 16, NHEAD, BATCH), ba(32);
  lorentz_attn_kernel<<<ga, ba, 0, stream>>>(Qh, Ql, Kh, Kl, Vh, Vl,
                                             mask, ascale, abias, out);
}
